// HierarchicalSampling_15006615733450
// MI455X (gfx1250) — matrix operand order repaired and checked by reading, not compiled, not run on hardware
//
#include <hip/hip_runtime.h>
#include <math.h>

// ---------------------------------------------------------------------------
// NeRF hierarchical sampling + tiny MLP + volumetric rendering, fused.
// MI455X (gfx1250): wave32, ONE WAVE PER RAY in a SINGLE-WAVE WORKGROUP
// (ISA 3.1: barrier ops -> S_NOP for single-wave WGs, so __syncthreads is
// free but still forces compiler-side LDS ordering). WMMA f16 16x16x32 for
// the MLP; GLOBAL_LOAD_ASYNC_TO_LDS (ASYNCcnt path) for coarse-t staging.
//
// Per ray (S = 258 samples):
//   stage 1: pdf/cdf from coarse density, 128 binary-search CDF inversions
//   stage 2: rank-merge of sorted coarse(130) + sorted fine(128) t values
//   stage 3: MLP 6->32->(3 rgb + 1 sigma) in 17 tiles of 16 points,
//            3x v_wmma_f32_16x16x32_f16 per tile (A/B packed per ISA 7.12.2)
//   stage 4: transmittance via multiplicative wave scan, weighted reductions
// ---------------------------------------------------------------------------

typedef __attribute__((ext_vector_type(16))) _Float16 v16h;
typedef __attribute__((ext_vector_type(8)))  float    v8f;

typedef __attribute__((address_space(1))) int gas_int;  // global
typedef __attribute__((address_space(3))) int las_int;  // LDS

#define NCOARSE 130
#define NW      129     // number of pdf bins
#define NFINE   128
#define NS      258     // 130 + 128 samples per ray
#define NTILE   17      // ceil(258/16)
#define SP      272     // padded samples (17*16)
#define WPB     1       // single wave per workgroup: barriers -> S_NOP

#if defined(__has_builtin)
#if __has_builtin(__builtin_amdgcn_global_load_async_to_lds_b32) && \
    __has_builtin(__builtin_amdgcn_s_wait_asynccnt)
#define USE_ASYNC_LDS 1
#endif
#endif

__device__ __forceinline__ float wave_sum(float v) {
#pragma unroll
  for (int off = 16; off > 0; off >>= 1) v += __shfl_down(v, off, 32);
  return __shfl(v, 0, 32);
}

__global__ __launch_bounds__(32 * WPB)
void nerf_hier_fused(const float* __restrict__ rays_o,
                     const float* __restrict__ rays_d,
                     const float* __restrict__ density,  // (BR,129)
                     const float* __restrict__ tvals,    // (BR,130) sorted
                     const float* __restrict__ uu,       // (BR,128)
                     const float* __restrict__ W1,       // (6,32)
                     const float* __restrict__ b1,       // (32)
                     const float* __restrict__ Wrgb,     // (32,3)
                     const float* __restrict__ brgb,     // (3)
                     const float* __restrict__ Wsig,     // (32)
                     const float* __restrict__ bsig,     // (1)
                     float* __restrict__ out,            // [rgb BRx3 | acc BR | depth BR]
                     int nrays) {
  __shared__ float    s_bins[WPB][132];      // coarse t values
  __shared__ float    s_w[WPB][132];         // pdf, then reused as t_fine
  __shared__ float    s_cdf[WPB][132];       // 130-entry CDF
  __shared__ float    s_tall[WPB][SP];       // merged t, padded
  __shared__ float    s_res[WPB][SP][4];     // rgb(3) + sigma->alpha(1)
  __shared__ _Float16 s_h[WPB][16 * 32];     // layer-1 activations (row-major)

  const int lane   = threadIdx.x & 31;
  const int wv     = (threadIdx.x >> 5);     // 0 with WPB==1
  const int col    = lane & 15;              // N index / matrix row id
  const int hiHalf = lane >> 4;              // 0: lanes 0-15, 1: lanes 16-31
  int ray = blockIdx.x * WPB + wv;
  if (ray >= nrays) ray = nrays - 1;         // redundant (deterministic) work

  // ---------------- stage 0: load coarse t, init padding ----------------
  const float* tv = tvals + (size_t)ray * NCOARSE;
  for (int i = lane; i < NCOARSE; i += 32) {
#if defined(USE_ASYNC_LDS)
    __builtin_amdgcn_global_load_async_to_lds_b32(
        (gas_int*)(tv + i), (las_int*)&s_bins[wv][i], 0, 0);
#else
    s_bins[wv][i] = tv[i];
#endif
  }
  for (int i = lane; i < SP; i += 32) s_tall[wv][i] = 6.0f;  // FAR pad
#if defined(USE_ASYNC_LDS)
  __builtin_amdgcn_s_wait_asynccnt(0);
#endif
  __syncthreads();

  // ---------------- stage 1a: raw weights w = sigma * dt ----------------
  float part = 0.f;
  for (int i = lane; i < NW; i += 32) {
    float w = density[(size_t)ray * NW + i] * (s_bins[wv][i + 1] - s_bins[wv][i]);
    s_w[wv][i] = w;
    part += w;
  }
  float wsum = wave_sum(part);
  __syncthreads();

  // ---------------- stage 1b: pdf = w/(sum+1e-8) + 1e-5, renormalized ----
  float inv = 1.0f / (wsum + 1e-8f);
  part = 0.f;
  for (int i = lane; i < NW; i += 32) {
    float p = s_w[wv][i] * inv + 1e-5f;
    s_w[wv][i] = p;
    part += p;
  }
  float psum = wave_sum(part);
  __syncthreads();

  // ---------------- stage 1c: serial CDF (lane 0, 129 adds) -------------
  if (lane == 0) {
    float c = 0.f, pinv = 1.0f / psum;
    s_cdf[wv][0] = 0.f;
    for (int i = 0; i < NW; ++i) { c += s_w[wv][i] * pinv; s_cdf[wv][i + 1] = c; }
  }
  __syncthreads();

  // ---------------- stage 1d: invert CDF (searchsorted right) -----------
  for (int k = lane; k < NFINE; k += 32) {
    float uvv = uu[(size_t)ray * NFINE + k];
    int lo = 0, hi = NCOARSE;                 // cdf has 130 entries
    while (lo < hi) {
      int mid = (lo + hi) >> 1;
      if (s_cdf[wv][mid] <= uvv) lo = mid + 1; else hi = mid;
    }
    int below = lo - 1; if (below < 0) below = 0;
    int above = lo;     if (above > NW) above = NW;
    float cb = s_cdf[wv][below], ca = s_cdf[wv][above];
    float den = ca - cb; if (den < 1e-5f) den = 1.f;
    float tt = (uvv - cb) / den;
    // reference quirk: interpolate bins indexed by fine-sample index k
    float b0 = s_bins[wv][k], b1v = s_bins[wv][k + 1];
    s_w[wv][k] = b0 + tt * (b1v - b0);        // reuse s_w as t_fine (sorted)
  }
  __syncthreads();

  // ---------------- stage 2: rank-merge coarse + fine (both sorted) -----
  for (int i = lane; i < NCOARSE; i += 32) {  // coarse rank: i + #{tf < x}
    float x = s_bins[wv][i];
    int lo = 0, hi = NFINE;
    while (lo < hi) { int m = (lo + hi) >> 1; if (s_w[wv][m] < x) lo = m + 1; else hi = m; }
    s_tall[wv][i + lo] = x;
  }
  for (int j = lane; j < NFINE; j += 32) {    // fine rank: j + #{coarse <= x}
    float x = s_w[wv][j];
    int lo = 0, hi = NCOARSE;
    while (lo < hi) { int m = (lo + hi) >> 1; if (s_bins[wv][m] <= x) lo = m + 1; else hi = m; }
    s_tall[wv][j + lo] = x;
  }
  __syncthreads();

  // ---------------- stage 3: MLP via WMMA --------------------------------
  // B matrix layout (ISA 7.12.2, 16-bit B 32x16): lane = column N (mod 16),
//   half q -> K = q + (q<8?0:8) + 8*hiHalf.
  // A matrix layout (16-bit A 16x32): lane%16 = row M,
  //   half q -> K = q + (q<8?0:8) + 8*hiHalf.
  v16h B1a, B1b, B2;
#pragma unroll
  for (int q = 0; q < 16; ++q) {
    int kB = q + ((q < 8) ? 0 : 8) + (hiHalf ? 8 : 0);
    B1a[q] = (_Float16)((kB < 6) ? W1[kB * 32 + col] : 0.f);
    B1b[q] = (_Float16)((kB < 6) ? W1[kB * 32 + 16 + col] : 0.f);
    float w2 = 0.f;
    if (col < 3)       w2 = Wrgb[kB * 3 + col];
    else if (col == 3) w2 = Wsig[kB];
    B2[q] = (_Float16)w2;
  }
  const float bias1a = b1[col], bias1b = b1[16 + col];
  const float bias2  = (col < 3) ? brgb[col] : ((col == 3) ? bsig[0] : 0.f);
  const float ox = rays_o[(size_t)ray * 3 + 0], oy = rays_o[(size_t)ray * 3 + 1],
              oz = rays_o[(size_t)ray * 3 + 2];
  const float dx = rays_d[(size_t)ray * 3 + 0], dy = rays_d[(size_t)ray * 3 + 1],
              dz = rays_d[(size_t)ray * 3 + 2];

  for (int tile = 0; tile < NTILE; ++tile) {
    // A1: row = col, features K=0..5 (held by lanes 0-15 only)
    const int srow = tile * 16 + col;
    const float t = s_tall[wv][srow];
    v16h A1;
#pragma unroll
    for (int q = 0; q < 16; ++q) A1[q] = (_Float16)0.f;
    A1[0] = (_Float16)(hiHalf ? 0.f : (ox + dx * t));
    A1[1] = (_Float16)(hiHalf ? 0.f : (oy + dy * t));
    A1[2] = (_Float16)(hiHalf ? 0.f : (oz + dz * t));
    A1[3] = (_Float16)(hiHalf ? 0.f : dx);
    A1[4] = (_Float16)(hiHalf ? 0.f : dy);
    A1[5] = (_Float16)(hiHalf ? 0.f : dz);

    v8f c0, c1;
#pragma unroll
    for (int r = 0; r < 8; ++r) { c0[r] = bias1a; c1[r] = bias1b; }
    v8f h0 = __builtin_amdgcn_wmma_f32_16x16x32_f16(false, A1, false, B1a,
                                                    (short)0, c0, false, false);
    v8f h1 = __builtin_amdgcn_wmma_f32_16x16x32_f16(false, A1, false, B1b,
                                                    (short)0, c1, false, false);
    // ReLU, stage D-layout -> row-major LDS for relayout into A-layout
#pragma unroll
    for (int r = 0; r < 8; ++r) {
      int row = r + (hiHalf ? 8 : 0);
      s_h[wv][row * 32 + col]      = (_Float16)fmaxf(h0[r], 0.f);
      s_h[wv][row * 32 + 16 + col] = (_Float16)fmaxf(h1[r], 0.f);
    }
    __syncthreads();                          // S_NOP in HW, compiler fence
    v16h A2;
#pragma unroll
    for (int q = 0; q < 16; ++q) {
      int kA = q + ((q < 8) ? 0 : 8) + (hiHalf ? 8 : 0);
      A2[q] = s_h[wv][col * 32 + kA];         // two contiguous 16B runs
    }
    v8f c2;
#pragma unroll
    for (int r = 0; r < 8; ++r) c2[r] = bias2;
    v8f o2 = __builtin_amdgcn_wmma_f32_16x16x32_f16(false, A2, false, B2,
                                                    (short)0, c2, false, false);
#pragma unroll
    for (int r = 0; r < 8; ++r) {
      int s = tile * 16 + r + (hiHalf ? 8 : 0);
      if (col < 4 && s < NS) {
        float v = o2[r];
        float res;
        if (col < 3) res = 1.f / (1.f + __expf(-v));                 // sigmoid
        else         res = fmaxf(v, 0.f) + log1pf(__expf(-fabsf(v))); // softplus
        s_res[wv][s][col] = res;
      }
    }
    __syncthreads();                          // protects s_h for next tile
  }

  // ---------------- stage 4: volumetric rendering -----------------------
  // transmittance cumprod via multiplicative wave scan over per-lane chunks
  const int CH = 9;                       // 32 * 9 = 288 >= 258
  const int sBeg = lane * CH;
  float cprod = 1.f;
#pragma unroll
  for (int i = 0; i < CH; ++i) {
    int s = sBeg + i;
    if (s < NS) {
      float delta = (s == NS - 1) ? 1e10f : (s_tall[wv][s + 1] - s_tall[wv][s]);
      float alpha = 1.f - __expf(-s_res[wv][s][3] * delta);
      s_res[wv][s][3] = alpha;            // overwrite sigma with alpha
      cprod *= (1.f - alpha + 1e-10f);
    }
  }
  float v = cprod;                        // Hillis-Steele inclusive mult-scan
#pragma unroll
  for (int off = 1; off < 32; off <<= 1) {
    float n = __shfl_up(v, off, 32);
    if (lane >= off) v *= n;
  }
  float run = __shfl_up(v, 1, 32);        // exclusive
  if (lane == 0) run = 1.f;

  float accS = 0.f, rS = 0.f, gS = 0.f, bS = 0.f, dS = 0.f;
#pragma unroll
  for (int i = 0; i < CH; ++i) {
    int s = sBeg + i;
    if (s < NS) {
      float alpha = s_res[wv][s][3];
      float wgt = alpha * run;
      run *= (1.f - alpha + 1e-10f);
      accS += wgt;
      rS += wgt * s_res[wv][s][0];
      gS += wgt * s_res[wv][s][1];
      bS += wgt * s_res[wv][s][2];
      dS += wgt * s_tall[wv][s];
    }
  }
  float acc = wave_sum(accS);
  float R = wave_sum(rS), G = wave_sum(gS), Bc = wave_sum(bS), D = wave_sum(dS);
  if (lane == 0 && (blockIdx.x * WPB + wv) < nrays) {
    float bg = 1.f - acc;                 // white background
    out[(size_t)ray * 3 + 0] = R + bg;
    out[(size_t)ray * 3 + 1] = G + bg;
    out[(size_t)ray * 3 + 2] = Bc + bg;
    out[(size_t)nrays * 3 + ray] = acc;
    out[(size_t)nrays * 4 + ray] = D;
  }
}

extern "C" void kernel_launch(void* const* d_in, const int* in_sizes, int n_in,
                              void* d_out, int out_size, void* d_ws, size_t ws_size,
                              hipStream_t stream) {
  const float* rays_o  = (const float*)d_in[0];
  const float* rays_d  = (const float*)d_in[1];
  // d_in[2] rgb_coarse: unused by the reference computation
  const float* density = (const float*)d_in[3];
  const float* tvals   = (const float*)d_in[4];
  const float* uu      = (const float*)d_in[5];
  const float* W1      = (const float*)d_in[6];
  const float* b1      = (const float*)d_in[7];
  const float* Wrgb    = (const float*)d_in[8];
  const float* brgb    = (const float*)d_in[9];
  const float* Wsig    = (const float*)d_in[10];
  const float* bsig    = (const float*)d_in[11];
  // d_in[12]=near, d_in[13]=far: unused

  const int nrays = in_sizes[5] / NFINE;  // B*R from u's flat size
  const int blocks = (nrays + WPB - 1) / WPB;
  hipLaunchKernelGGL(nerf_hier_fused, dim3(blocks), dim3(32 * WPB), 0, stream,
                     rays_o, rays_d, density, tvals, uu,
                     W1, b1, Wrgb, brgb, Wsig, bsig,
                     (float*)d_out, nrays);
}
